// GroupQueryAttention_12352325943684
// MI455X (gfx1250) — matrix operand order repaired and checked by reading, not compiled, not run on hardware
//
#include <hip/hip_runtime.h>
#include <hip/hip_bf16.h>

typedef __attribute__((ext_vector_type(16))) _Float16 v16h;
typedef __attribute__((ext_vector_type(8)))  _Float16 v8h;
typedef __attribute__((ext_vector_type(8)))  float    v8f;
typedef __attribute__((ext_vector_type(4)))  float    v4f;
typedef __attribute__((__vector_size__(16))) int      v4i;

// ---------------- CDNA5 feature guards (compile-safe fallbacks) -------------
#if defined(__has_builtin)
#if __has_builtin(__builtin_amdgcn_global_load_async_to_lds_b128) && \
    __has_builtin(__builtin_amdgcn_s_wait_asynccnt)
#define HAVE_ASYNC_LDS 1
#endif
#if __has_builtin(__builtin_amdgcn_tanhf)
#define FAST_TANH(x) __builtin_amdgcn_tanhf(x)
#endif
#endif
#ifndef HAVE_ASYNC_LDS
#define HAVE_ASYNC_LDS 0
#endif
#ifndef FAST_TANH
#define FAST_TANH(x) tanhf(x)
#endif

// 16-byte global -> LDS copy, async if the gfx1250 builtin is available.
__device__ __forceinline__ void copy16_g2l(const _Float16* g, _Float16* l) {
#if HAVE_ASYNC_LDS
    __builtin_amdgcn_global_load_async_to_lds_b128(
        (__attribute__((address_space(1))) v4i*)const_cast<_Float16*>(g),
        (__attribute__((address_space(3))) v4i*)l, 0, 0);
#else
    *(v8h*)l = *(const v8h*)g;
#endif
}
__device__ __forceinline__ void copy16_wait() {
#if HAVE_ASYNC_LDS
    __builtin_amdgcn_s_wait_asynccnt(0);
#endif
}

// Build a 16-element f16 fragment from two 16-byte chunks.
__device__ __forceinline__ v16h ld_frag(const _Float16* p0, const _Float16* p1) {
    v8h a = *(const v8h*)p0;
    v8h b = *(const v8h*)p1;
    v16h r;
#pragma unroll
    for (int i = 0; i < 8; ++i) { r[i] = a[i]; r[i + 8] = b[i]; }
    return r;
}

// ---------------------------------------------------------------------------
// Generic tiled GEMM: C[M,N] = A[M,K] @ B[K,N]
//   A: fp32 or f16 (A_HALF), row-major, lda=K
//   B: fp32 row-major, ldb=N
//   C: f16 or fp32 (C_HALF); C_TRANS_BT stores C transposed per 256-row group:
//      C[(r/256)*N + c][r%256]  (used for the V projection -> vT layout)
// Block tile 64x64, 256 threads (8 waves). Wave w: row-tile rt=w&3,
// col-half ch=w>>2 (two 16-col tiles). K-step = 64 (two WMMA K-chunks).
// ---------------------------------------------------------------------------
template <bool A_HALF, bool C_HALF, bool C_TRANS_BT>
__global__ __launch_bounds__(256) void gemm_wmma_kernel(
    const void* __restrict__ Aptr, const float* __restrict__ Bptr,
    void* __restrict__ Cptr, int M, int N, int K) {
    __shared__ _Float16 As[64 * 72];   // [row][k] padded stride 72
    __shared__ _Float16 Bt[64 * 72];   // [col][k] (B transposed) stride 72

    const int tid  = threadIdx.x;
    const int lane = tid & 31;
    const int w    = tid >> 5;
    const int hl   = (lane >> 4) & 1;
    const int l15  = lane & 15;
    const int rt   = w & 3;      // row tile within block
    const int ch   = w >> 2;     // column half (0/1)
    const int row0 = blockIdx.y * 64;
    const int col0 = blockIdx.x * 64;

    // A-tile loader: thread -> (row ar, 16 consecutive k at akq)
    const int ar  = tid >> 2;           // 0..63
    const int akq = (tid & 3) * 16;     // 0,16,32,48
    // B-tile loader: thread -> (k rows bk,bk+32; 8 consecutive n at bn)
    const int bk = tid >> 3;            // 0..31
    const int bn = (tid & 7) * 8;       // 0..56

    v8f acc0 = {};
    v8f acc1 = {};

    for (int kk = 0; kk < K; kk += 64) {
        __syncthreads();
        // ---- stage A tile (convert to f16 if needed) ----
        if constexpr (A_HALF) {
            const _Float16* A = (const _Float16*)Aptr;
            const _Float16* p = A + (size_t)(row0 + ar) * K + kk + akq;
            *(v8h*)(As + ar * 72 + akq)     = *(const v8h*)p;
            *(v8h*)(As + ar * 72 + akq + 8) = *(const v8h*)(p + 8);
        } else {
            const float* A = (const float*)Aptr;
            const float* p = A + (size_t)(row0 + ar) * K + kk + akq;
#pragma unroll
            for (int half = 0; half < 2; ++half) {
                v4f x0 = *(const v4f*)(p + half * 8);
                v4f x1 = *(const v4f*)(p + half * 8 + 4);
                v8h av;
#pragma unroll
                for (int i = 0; i < 4; ++i) {
                    av[i]     = (_Float16)x0[i];
                    av[i + 4] = (_Float16)x1[i];
                }
                *(v8h*)(As + ar * 72 + akq + half * 8) = av;
            }
        }
        // ---- stage B tile transposed (two k-rows per thread) ----
#pragma unroll
        for (int s = 0; s < 2; ++s) {
            const int kb = bk + s * 32;
            const float* p = Bptr + (size_t)(kk + kb) * N + col0 + bn;
            v4f x0 = *(const v4f*)p;
            v4f x1 = *(const v4f*)(p + 4);
#pragma unroll
            for (int i = 0; i < 4; ++i) {
                Bt[(bn + i) * 72 + kb]     = (_Float16)x0[i];
                Bt[(bn + i + 4) * 72 + kb] = (_Float16)x1[i];
            }
        }
        __syncthreads();

        // ---- fragments + WMMA (two K-chunks of 32) ----
#pragma unroll
        for (int kc = 0; kc < 2; ++kc) {
            const _Float16* arow = As + (rt * 16 + l15) * 72 + kc * 32;
            v16h aF = ld_frag(arow + hl * 16, arow + hl * 16 + 8);

            const _Float16* b0 = Bt + (ch * 32 + l15) * 72 + kc * 32;
            v16h bF0 = ld_frag(b0 + hl * 16, b0 + hl * 16 + 8);
            const _Float16* b1 = Bt + (ch * 32 + 16 + l15) * 72 + kc * 32;
            v16h bF1 = ld_frag(b1 + hl * 16, b1 + hl * 16 + 8);

            acc0 = __builtin_amdgcn_wmma_f32_16x16x32_f16(false, aF, false, bF0,
                                                          (short)0, acc0, false, false);
            acc1 = __builtin_amdgcn_wmma_f32_16x16x32_f16(false, aF, false, bF1,
                                                          (short)0, acc1, false, false);
        }
    }

    // ---- store C (C layout: elem j of lane -> row j+8*hl, col l15) ----
#pragma unroll
    for (int j = 0; j < 8; ++j) {
        const int r  = row0 + rt * 16 + j + 8 * hl;
        const int c0 = col0 + ch * 32 + l15;
        if constexpr (C_TRANS_BT) {
            // vT layout: [bt][col][m], m = r%256, 256 rows per bt group
            _Float16* C = (_Float16*)Cptr;
            const size_t bt = (size_t)(r >> 8);
            const int   m  = r & 255;
            C[(bt * N + c0) * 256 + m]        = (_Float16)acc0[j];
            C[(bt * N + c0 + 16) * 256 + m]   = (_Float16)acc1[j];
        } else if constexpr (C_HALF) {
            _Float16* C = (_Float16*)Cptr;
            C[(size_t)r * N + c0]      = (_Float16)acc0[j];
            C[(size_t)r * N + c0 + 16] = (_Float16)acc1[j];
        } else {
            float* C = (float*)Cptr;
            C[(size_t)r * N + c0]      = acc0[j];
            C[(size_t)r * N + c0 + 16] = acc1[j];
        }
    }
}

// ---------------------------------------------------------------------------
// Attention: one block per (b*t, head). H=8, G=2, R=4, N=256, D=64.
// scores = Q @ K^T * 0.125 ; attn = tanh(scores) + sgr ; out = attn @ V
// 8 waves, wave w owns query rows [32w, 32w+32). K/V tiles staged to LDS with
// async global->LDS copies when available.
// ---------------------------------------------------------------------------
__global__ __launch_bounds__(256) void gqa_tanh_attn_kernel(
    const _Float16* __restrict__ qh,   // [BT*256, 512] (n, h*64+d)
    const _Float16* __restrict__ kh,   // [BT*256, 128] (m, g*64+d)
    const _Float16* __restrict__ vT,   // [BT, 128, 256] (g*64+d, m) transposed
    const float* __restrict__ sgr,     // [256, 256]
    _Float16* __restrict__ atto) {     // [BT*256, 512]
    __shared__ _Float16 Kt[32 * 64];        // [m_local][d]
    __shared__ _Float16 Vt[64 * 32];        // [d][m_local]
    __shared__ _Float16 Sc[8 * 32 * 32];    // per-wave score scratch [row][m]

    const int bh  = blockIdx.x;
    const int bt  = bh >> 3;
    const int hh  = bh & 7;
    const int g   = hh >> 2;          // kv group = head / R
    const int tid = threadIdx.x;
    const int lane = tid & 31;
    const int w    = tid >> 5;
    const int hl   = (lane >> 4) & 1;
    const int l15  = lane & 15;
    const int r0   = w * 32;          // this wave's query row base

    const _Float16* qbase  = qh + (size_t)bt * 256 * 512 + hh * 64;
    const _Float16* kbase  = kh + (size_t)bt * 256 * 128 + g * 64;
    const _Float16* vtbase = vT + ((size_t)bt * 128 + g * 64) * 256;

    // Preload Q fragments: aQ[rt][kc] covers rows rt*16.., d-chunk kc*32..
    v16h aQ[2][2];
#pragma unroll
    for (int rt = 0; rt < 2; ++rt) {
#pragma unroll
        for (int kc = 0; kc < 2; ++kc) {
            const _Float16* qr = qbase + (size_t)(r0 + rt * 16 + l15) * 512 + kc * 32;
            aQ[rt][kc] = ld_frag(qr + hl * 16, qr + hl * 16 + 8);
        }
    }

    v8f oacc[2][4];
#pragma unroll
    for (int rt = 0; rt < 2; ++rt)
#pragma unroll
        for (int dt = 0; dt < 4; ++dt) oacc[rt][dt] = (v8f){};

    // K tile loader: thread -> (m row lm, 8 consecutive d at ld8)
    const int lm  = tid >> 3;          // 0..31
    const int ld8 = (tid & 7) * 8;     // 0..56
    // V tile loader: thread -> (d row vd, 8 consecutive m at vp)
    const int vd = tid >> 2;           // 0..63
    const int vp = (tid & 3) * 8;      // 0,8,16,24

    _Float16* myS = Sc + w * 32 * 32;

    for (int m0 = 0; m0 < 256; m0 += 32) {
        __syncthreads();
        // stage K tile [32 m][64 d] and V tile [64 d][32 m] (already transposed)
        copy16_g2l(kbase + (size_t)(m0 + lm) * 128 + ld8, Kt + lm * 64 + ld8);
        copy16_g2l(vtbase + (size_t)vd * 256 + m0 + vp, Vt + vd * 32 + vp);
        copy16_wait();
        __syncthreads();

        // ---- scores = Q @ K^T over this 32-key tile ----
#pragma unroll
        for (int rt = 0; rt < 2; ++rt) {
#pragma unroll
            for (int mt = 0; mt < 2; ++mt) {
                v8f s = {};
#pragma unroll
                for (int kc = 0; kc < 2; ++kc) {
                    const _Float16* kr = Kt + (mt * 16 + l15) * 64 + kc * 32 + hl * 16;
                    v16h bK = ld_frag(kr, kr + 8);
                    s = __builtin_amdgcn_wmma_f32_16x16x32_f16(
                        false, aQ[rt][kc], false, bK, (short)0, s, false, false);
                }
                // tanh(s*scale) + sgr, write f16 to per-wave scratch
                const int colm = mt * 16 + l15;
#pragma unroll
                for (int j = 0; j < 8; ++j) {
                    const int rloc = rt * 16 + j + 8 * hl;
                    float a = FAST_TANH(s[j] * 0.125f) +
                              sgr[(size_t)(r0 + rloc) * 256 + m0 + colm];
                    myS[rloc * 32 + colm] = (_Float16)a;
                }
            }
        }

        // ---- out += attn_tile @ V_tile ----
#pragma unroll
        for (int rt = 0; rt < 2; ++rt) {
            const _Float16* sr = myS + (rt * 16 + l15) * 32;
            v16h aS = ld_frag(sr + hl * 16, sr + hl * 16 + 8);
#pragma unroll
            for (int dt = 0; dt < 4; ++dt) {
                const _Float16* vr = Vt + (dt * 16 + l15) * 32 + hl * 16;
                v16h bV = ld_frag(vr, vr + 8);
                oacc[rt][dt] = __builtin_amdgcn_wmma_f32_16x16x32_f16(
                    false, aS, false, bV, (short)0, oacc[rt][dt], false, false);
            }
        }
    }

    // ---- store attention output (f16) at [bt, n, hh*64 + d] ----
    _Float16* obase = atto + (size_t)bt * 256 * 512 + hh * 64;
#pragma unroll
    for (int rt = 0; rt < 2; ++rt) {
#pragma unroll
        for (int dt = 0; dt < 4; ++dt) {
#pragma unroll
            for (int j = 0; j < 8; ++j) {
                const int n = r0 + rt * 16 + j + 8 * hl;
                const int d = dt * 16 + l15;
                obase[(size_t)n * 512 + d] = (_Float16)oacc[rt][dt][j];
            }
        }
    }
}

// ---------------------------------------------------------------------------
// Host-side launcher
// ---------------------------------------------------------------------------
extern "C" void kernel_launch(void* const* d_in, const int* in_sizes, int n_in,
                              void* d_out, int out_size, void* d_ws, size_t ws_size,
                              hipStream_t stream) {
    (void)in_sizes; (void)n_in; (void)out_size; (void)ws_size;

    const float* x   = (const float*)d_in[0];  // [4,32,256,512] -> [32768,512]
    const float* sgr = (const float*)d_in[1];  // [256,256]
    const float* Wq  = (const float*)d_in[2];  // [512,512]
    const float* Wk  = (const float*)d_in[3];  // [512,128]
    const float* Wv  = (const float*)d_in[4];  // [512,128]
    const float* Wo  = (const float*)d_in[5];  // [512,512]
    float* out = (float*)d_out;                // [32768,512]

    const int M = 32768, C = 512, GD = 128;

    // Workspace layout (f16): q[32768*512], k[32768*128], vT[128*128*256], att[32768*512]
    _Float16* qh  = (_Float16*)d_ws;
    _Float16* kh  = qh + (size_t)M * C;
    _Float16* vTh = kh + (size_t)M * GD;
    _Float16* att = vTh + (size_t)M * GD;

    dim3 blk(256);

    // Projections: fp32 A, fp32 B, f16 C
    gemm_wmma_kernel<false, true, false><<<dim3(C / 64, M / 64), blk, 0, stream>>>(
        x, Wq, qh, M, C, C);
    gemm_wmma_kernel<false, true, false><<<dim3(GD / 64, M / 64), blk, 0, stream>>>(
        x, Wk, kh, M, GD, C);
    // V projection stored transposed per (b,t) group: vT[bt][g*64+d][m]
    gemm_wmma_kernel<false, true, true><<<dim3(GD / 64, M / 64), blk, 0, stream>>>(
        x, Wv, vTh, M, GD, C);

    // Attention: one block per (b*t, head) = 128*8 = 1024 blocks
    gqa_tanh_attn_kernel<<<dim3(1024), blk, 0, stream>>>(qh, kh, vTh, sgr, att);

    // Output projection: f16 A, fp32 B, fp32 C
    gemm_wmma_kernel<true, false, false><<<dim3(C / 64, M / 64), blk, 0, stream>>>(
        att, Wo, out, M, C, C);
}
